// Grapher_49211735277824
// MI455X (gfx1250) — matrix operand order repaired and checked by reading, not compiled, not run on hardware
//
#include <hip/hip_runtime.h>
#include <hip/hip_bf16.h>

// ---------------------------------------------------------------------------
// Grapher block for MI455X (gfx1250, wave32, WMMA 16x16x32 bf16).
// Edge-conv refactored:  y(n,k) = (W1-W2) x_n + W2 x_j  -> one 1536x384 GEMM
// + gather/max pass.  ep refactored: mean_k before w_up.  All big GEMMs run
// on v_wmma_f32_16x16x32_bf16 with layout-exact fragment loads.
// ---------------------------------------------------------------------------

typedef unsigned short u16;
typedef unsigned int   u32;
typedef __attribute__((ext_vector_type(8)))  float  v8f;
typedef __attribute__((ext_vector_type(16))) __bf16 v16bf;
typedef __attribute__((ext_vector_type(4)))  u32    u32x4;

#define BB   64
#define CC   384
#define C2   768
#define HWX  196
#define NN   210
#define NP   224
#define RR   32
#define KNN  9
#define EPS  1e-5f

__device__ __forceinline__ u16 f2bf(float f) {
  u32 u = __float_as_uint(f);
  u += 0x7FFFu + ((u >> 16) & 1u);
  return (u16)(u >> 16);
}
__device__ __forceinline__ float bf2f(u16 h) { return __uint_as_float(((u32)h) << 16); }

union Frag {
  v16bf v;
  u16   u[16];
  u32x4 q[2];
};

struct Tile32 { v8f c[2][2]; };

__device__ __forceinline__ void tile_zero(Tile32& t) {
  v8f z = {0.f,0.f,0.f,0.f,0.f,0.f,0.f,0.f};
  t.c[0][0] = z; t.c[0][1] = z; t.c[1][0] = z; t.c[1][1] = z;
}

// A: row-major MxK (lda), B: transposed activations, row-major NxK (ldb).
// Fragment layouts per CDNA5 ISA 7.12.2 (16-bit A 16x32, B 32x16, wave32).
__device__ __forceinline__ void gemm_tile(const u16* __restrict__ A,
                                          const u16* __restrict__ Bp,
                                          int m0, int n0, int K, int lda, int ldb,
                                          int lane, Tile32& t) {
  const int hf  = lane >> 4;
  const int l16 = lane & 15;
  for (int k0 = 0; k0 < K; k0 += 32) {
    Frag a0, a1, b0, b1;
    {
      const u16* pa = A + (size_t)(m0 + l16) * lda + k0 + hf * 8;
      a0.q[0] = *(const u32x4*)(pa);
      a0.q[1] = *(const u32x4*)(pa + 16);
      pa += (size_t)16 * lda;
      a1.q[0] = *(const u32x4*)(pa);
      a1.q[1] = *(const u32x4*)(pa + 16);
    }
    {
      const u16* pb = Bp + (size_t)(n0 + l16) * ldb + k0 + hf * 8;
      b0.q[0] = *(const u32x4*)(pb);
      b0.q[1] = *(const u32x4*)(pb + 16);
      pb += (size_t)16 * ldb;
      b1.q[0] = *(const u32x4*)(pb);
      b1.q[1] = *(const u32x4*)(pb + 16);
      __builtin_prefetch(pb + 32, 0, 0);  // global_prefetch_b8 for next K slab
    }
    t.c[0][0] = __builtin_amdgcn_wmma_f32_16x16x32_bf16(false, a0.v, false, b0.v, (short)0, t.c[0][0], false, false);
    t.c[0][1] = __builtin_amdgcn_wmma_f32_16x16x32_bf16(false, a0.v, false, b1.v, (short)0, t.c[0][1], false, false);
    t.c[1][0] = __builtin_amdgcn_wmma_f32_16x16x32_bf16(false, a1.v, false, b0.v, (short)0, t.c[1][0], false, false);
    t.c[1][1] = __builtin_amdgcn_wmma_f32_16x16x32_bf16(false, a1.v, false, b1.v, (short)0, t.c[1][1], false, false);
  }
}

// ------------------------- weight prep (BN folding) -------------------------

__global__ void k_prep_w1(const float* __restrict__ w, const float* __restrict__ b,
                          const float* __restrict__ g, const float* __restrict__ bt,
                          const float* __restrict__ m, const float* __restrict__ v,
                          u16* __restrict__ wbf, float* __restrict__ bias) {
  int i = blockIdx.x * blockDim.x + threadIdx.x;
  if (i >= CC * CC) return;
  int o = i / CC, c = i % CC;
  float s = g[o] / sqrtf(v[o] + EPS);
  wbf[i] = f2bf(w[i] * s);
  if (c == 0) bias[o] = (b[o] - m[o]) * s + bt[o];
}

__global__ void k_prep_small(const float* __restrict__ wdown, const float* __restrict__ gp,
                             u16* __restrict__ wdownbf, u16* __restrict__ gptbf) {
  int i = blockIdx.x * blockDim.x + threadIdx.x;
  if (i < RR * CC) {
    wdownbf[i] = f2bf(wdown[i]);
  } else if (i < 2 * RR * CC) {
    int j = i - RR * CC;
    int c = j / RR, r = j % RR;
    gptbf[c * RR + r] = f2bf(gp[r * CC + c]);   // (C x R) = graph_prompt^T
  }
}

__global__ void k_prep_wcomb(const float* __restrict__ wec, const float* __restrict__ bec,
                             const float* __restrict__ g, const float* __restrict__ bt,
                             const float* __restrict__ m, const float* __restrict__ v,
                             u16* __restrict__ wcomb, float* __restrict__ se, float* __restrict__ ce) {
  int i = blockIdx.x * blockDim.x + threadIdx.x;
  if (i >= 2 * C2 * CC) return;
  int o = i / CC, c = i % CC;
  float val;
  if (o < C2) val = wec[o * C2 + c] - wec[o * C2 + CC + c];   // W1 - W2
  else        val = wec[(o - C2) * C2 + CC + c];              // W2
  wcomb[i] = f2bf(val);
  if (c == 0 && o < C2) {
    float s = g[o] / sqrtf(v[o] + EPS);
    se[o] = s;
    ce[o] = (bec[o] - m[o]) * s + bt[o];
  }
}

__global__ void k_prep_w2(const float* __restrict__ w, const float* __restrict__ b,
                          const float* __restrict__ g, const float* __restrict__ bt,
                          const float* __restrict__ m, const float* __restrict__ v,
                          u16* __restrict__ wbf, float* __restrict__ bias) {
  int i = blockIdx.x * blockDim.x + threadIdx.x;
  if (i >= CC * C2) return;
  int c = i / C2, o = i % C2;
  float s = g[c] / sqrtf(v[c] + EPS);
  wbf[i] = f2bf(w[i] * s);
  if (o == 0) bias[c] = (b[c] - m[c]) * s + bt[c];
}

// ------------------------- activation staging -------------------------

__global__ void k_x_to_bf(const float* __restrict__ x, u16* __restrict__ xT) {
  int i = blockIdx.x * blockDim.x + threadIdx.x;
  if (i >= BB * NP * CC) return;
  int c = i % CC, n = (i / CC) % NP, b = i / (CC * NP);
  float v = (n < HWX) ? x[((size_t)b * CC + c) * HWX + n] : 0.f;
  xT[i] = f2bf(v);
}

__global__ void k_prompt_pad(const float* __restrict__ npr, float* __restrict__ h) {
  int i = blockIdx.x * blockDim.x + threadIdx.x;
  if (i >= BB * CC * 28) return;
  int j = i % 28, c = (i / 28) % CC, b = i / (28 * CC);
  float v = (j < 14) ? npr[c * 14 + j] : 0.f;
  h[((size_t)b * CC + c) * NP + (HWX + j)] = v;
}

__global__ void k_h_to_bf(const float* __restrict__ h, u16* __restrict__ hT) {
  int i = blockIdx.x * blockDim.x + threadIdx.x;
  if (i >= BB * NP * CC) return;
  int c = i % CC, n = (i / CC) % NP, b = i / (CC * NP);
  hT[i] = f2bf(h[((size_t)b * CC + c) * NP + n]);
}

// ------------------------- WMMA GEMM kernels -------------------------

__global__ void k_gemm_fc1(const u16* __restrict__ Wbf, const u16* __restrict__ xT,
                           const float* __restrict__ bias, float* __restrict__ h) {
  int b = blockIdx.z, wv = threadIdx.x >> 5, lane = threadIdx.x & 31;
  int m0 = blockIdx.x * 256 + wv * 32, n0 = blockIdx.y * 32;
  if (m0 >= CC) return;
  Tile32 t; tile_zero(t);
  gemm_tile(Wbf, xT + (size_t)b * NP * CC, m0, n0, CC, CC, CC, lane, t);
  int hf = lane >> 4, l16 = lane & 15;
#pragma unroll
  for (int mi = 0; mi < 2; ++mi)
#pragma unroll
    for (int ni = 0; ni < 2; ++ni)
#pragma unroll
      for (int r = 0; r < 8; ++r) {
        int m = m0 + mi * 16 + hf * 8 + r, n = n0 + ni * 16 + l16;
        if (n < HWX) h[((size_t)b * CC + m) * NP + n] = t.c[mi][ni][r] + bias[m];
      }
}

__global__ void k_gemm_down(const u16* __restrict__ Wbf, const u16* __restrict__ hT,
                            const float* __restrict__ bdown,
                            float* __restrict__ lr, u16* __restrict__ lrT) {
  int b = blockIdx.z, wv = threadIdx.x >> 5, lane = threadIdx.x & 31;
  int n0 = wv * 32;
  if (n0 >= NP) return;
  Tile32 t; tile_zero(t);
  gemm_tile(Wbf, hT + (size_t)b * NP * CC, 0, n0, CC, CC, CC, lane, t);
  int hf = lane >> 4, l16 = lane & 15;
#pragma unroll
  for (int mi = 0; mi < 2; ++mi)
#pragma unroll
    for (int ni = 0; ni < 2; ++ni)
#pragma unroll
      for (int r = 0; r < 8; ++r) {
        int m = mi * 16 + hf * 8 + r, n = n0 + ni * 16 + l16;
        float val = t.c[mi][ni][r] + bdown[m];
        float gl = 0.5f * val * (1.f + erff(val * 0.70710678f));
        if (n >= NN) gl = 0.f;
        lr[((size_t)b * RR + m) * NP + n] = gl;
        lrT[((size_t)b * NP + n) * RR + m] = f2bf(gl);
      }
}

__global__ void k_gemm_res(const u16* __restrict__ gpT, const u16* __restrict__ lrT,
                           const float* __restrict__ h, u16* __restrict__ xfT) {
  int b = blockIdx.z, wv = threadIdx.x >> 5, lane = threadIdx.x & 31;
  int m0 = blockIdx.x * 256 + wv * 32, n0 = blockIdx.y * 32;
  if (m0 >= CC) return;
  Tile32 t; tile_zero(t);
  gemm_tile(gpT, lrT + (size_t)b * NP * RR, m0, n0, RR, RR, RR, lane, t);
  int hf = lane >> 4, l16 = lane & 15;
#pragma unroll
  for (int mi = 0; mi < 2; ++mi)
#pragma unroll
    for (int ni = 0; ni < 2; ++ni)
#pragma unroll
      for (int r = 0; r < 8; ++r) {
        int m = m0 + mi * 16 + hf * 8 + r, n = n0 + ni * 16 + l16;
        float xv = 0.8f * h[((size_t)b * CC + m) * NP + n] + 0.2f * t.c[mi][ni][r];
        xfT[((size_t)b * NP + n) * CC + m] = f2bf(xv);
      }
}

__global__ void k_norm(const u16* __restrict__ xfT, float* __restrict__ nrm) {
  int wv = threadIdx.x >> 5, lane = threadIdx.x & 31;
  int row = blockIdx.x * 8 + wv;
  if (row >= BB * NP) return;
  const u16* p = xfT + (size_t)row * CC;
  float s = 0.f;
  for (int c = lane; c < CC; c += 32) { float v = bf2f(p[c]); s += v * v; }
  s += __shfl_xor(s, 16); s += __shfl_xor(s, 8); s += __shfl_xor(s, 4);
  s += __shfl_xor(s, 2);  s += __shfl_xor(s, 1);
  if (lane == 0) nrm[row] = sqrtf(s) + 1e-12f;
}

__global__ void k_gemm_gram(const u16* __restrict__ xfT, float* __restrict__ G) {
  int b = blockIdx.z, wv = threadIdx.x >> 5, lane = threadIdx.x & 31;
  int m0 = wv * 32, n0 = blockIdx.y * 32;
  if (m0 >= NP) return;
  const u16* X = xfT + (size_t)b * NP * CC;
  Tile32 t; tile_zero(t);
  gemm_tile(X, X, m0, n0, CC, CC, CC, lane, t);
  int hf = lane >> 4, l16 = lane & 15;
#pragma unroll
  for (int mi = 0; mi < 2; ++mi)
#pragma unroll
    for (int ni = 0; ni < 2; ++ni)
#pragma unroll
      for (int r = 0; r < 8; ++r) {
        int m = m0 + mi * 16 + hf * 8 + r, n = n0 + ni * 16 + l16;
        G[((size_t)b * NP + m) * NP + n] = t.c[mi][ni][r];
      }
}

__global__ void k_topk(const float* __restrict__ G, const float* __restrict__ nrm,
                       int* __restrict__ nn) {
  int i = blockIdx.x * blockDim.x + threadIdx.x;
  if (i >= BB * NN) return;
  int n = i % NN, b = i / NN;
  float nm = nrm[b * NP + n];
  const float* Gr = G + ((size_t)b * NP + n) * NP;
  float bv[KNN]; int bi[KNN];
#pragma unroll
  for (int j = 0; j < KNN; ++j) { bv[j] = -1e30f; bi[j] = 0; }
  for (int m = 0; m < NN; ++m) {
    float v = Gr[m] / (nm * nrm[b * NP + m]);
    int id = m;
#pragma unroll
    for (int j = 0; j < KNN; ++j) {
      if (v > bv[j]) { float tv = bv[j]; int ti = bi[j]; bv[j] = v; bi[j] = id; v = tv; id = ti; }
    }
  }
  int* o = nn + ((size_t)b * NN + n) * KNN;
#pragma unroll
  for (int j = 0; j < KNN; ++j) o[j] = bi[j];
}

__global__ void k_gemm_big(const u16* __restrict__ wcomb, const u16* __restrict__ xfT,
                           float* __restrict__ Z) {
  int b = blockIdx.z, wv = threadIdx.x >> 5, lane = threadIdx.x & 31;
  int m0 = blockIdx.x * 256 + wv * 32, n0 = blockIdx.y * 32;
  if (m0 >= 2 * C2) return;
  Tile32 t; tile_zero(t);
  gemm_tile(wcomb, xfT + (size_t)b * NP * CC, m0, n0, CC, CC, CC, lane, t);
  int hf = lane >> 4, l16 = lane & 15;
#pragma unroll
  for (int mi = 0; mi < 2; ++mi)
#pragma unroll
    for (int ni = 0; ni < 2; ++ni)
#pragma unroll
      for (int r = 0; r < 8; ++r) {
        int m = m0 + mi * 16 + hf * 8 + r, n = n0 + ni * 16 + l16;
        Z[((size_t)b * (2 * C2) + m) * NP + n] = t.c[mi][ni][r];
      }
}

__global__ void k_edge_max(const float* __restrict__ Z, const int* __restrict__ nn,
                           const float* __restrict__ se, const float* __restrict__ ce,
                           u16* __restrict__ ymT) {
  int i = blockIdx.x * blockDim.x + threadIdx.x;
  if (i >= BB * NP * C2) return;
  int o = i % C2, n = (i / C2) % NP, b = i / (C2 * NP);
  float best = 0.f;
  if (n < NN) {
    const float* Za = Z + ((size_t)b * (2 * C2) + o) * NP;
    const float* Zb = Z + ((size_t)b * (2 * C2) + C2 + o) * NP;
    float a0 = Za[n], s = se[o], c0 = ce[o];
    const int* idx = nn + ((size_t)b * NN + n) * KNN;
#pragma unroll
    for (int k = 0; k < KNN; ++k) {
      float v = (a0 + Zb[idx[k]]) * s + c0;   // relu via best>=0 init
      best = fmaxf(best, v);
    }
  }
  ymT[i] = f2bf(best);
}

__global__ void k_lrmean(const float* __restrict__ lr, const int* __restrict__ nn,
                         float* __restrict__ lrmT) {
  int i = blockIdx.x * blockDim.x + threadIdx.x;
  if (i >= BB * NP * RR) return;
  int r = i % RR, n = (i / RR) % NP, b = i / (RR * NP);
  float s = 0.f;
  if (n < NN) {
    const int* idx = nn + ((size_t)b * NN + n) * KNN;
    const float* lrow = lr + ((size_t)b * RR + r) * NP;
#pragma unroll
    for (int k = 0; k < KNN; ++k) s += lrow[idx[k]];
    s *= (1.f / 9.f);
  }
  lrmT[i] = s;
}

__global__ void k_gemm_fc2(const u16* __restrict__ w2bf, const u16* __restrict__ ymT,
                           const float* __restrict__ bias2, const float* __restrict__ wup,
                           const float* __restrict__ bup, const float* __restrict__ lrmT,
                           const float* __restrict__ x, float* __restrict__ out) {
  int b = blockIdx.z, wv = threadIdx.x >> 5, lane = threadIdx.x & 31;
  int m0 = blockIdx.x * 256 + wv * 32, n0 = blockIdx.y * 32;
  if (m0 >= CC) return;
  Tile32 t; tile_zero(t);
  gemm_tile(w2bf, ymT + (size_t)b * NP * C2, m0, n0, C2, C2, C2, lane, t);
  int hf = lane >> 4, l16 = lane & 15;
#pragma unroll
  for (int mi = 0; mi < 2; ++mi)
#pragma unroll
    for (int ni = 0; ni < 2; ++ni)
#pragma unroll
      for (int r = 0; r < 8; ++r) {
        int m = m0 + mi * 16 + hf * 8 + r, n = n0 + ni * 16 + l16;
        if (n < HWX) {
          float ep = bup[m];
          const float* lv = lrmT + ((size_t)b * NP + n) * RR;
          const float* wu = wup + m * RR;
#pragma unroll
          for (int q = 0; q < RR; ++q) ep += wu[q] * lv[q];
          size_t oi = ((size_t)b * CC + m) * HWX + n;
          out[oi] = 0.8f * (t.c[mi][ni][r] + bias2[m]) + 0.2f * ep + x[oi];
        }
      }
}

// ------------------------- host launch -------------------------

extern "C" void kernel_launch(void* const* d_in, const int* in_sizes, int n_in,
                              void* d_out, int out_size, void* d_ws, size_t ws_size,
                              hipStream_t stream) {
  (void)in_sizes; (void)n_in; (void)out_size; (void)ws_size;
  const float* x      = (const float*)d_in[0];
  const float* w_fc1  = (const float*)d_in[1];
  const float* b_fc1  = (const float*)d_in[2];
  const float* bn1_g  = (const float*)d_in[3];
  const float* bn1_b  = (const float*)d_in[4];
  const float* bn1_m  = (const float*)d_in[5];
  const float* bn1_v  = (const float*)d_in[6];
  const float* nprom  = (const float*)d_in[7];
  const float* w_down = (const float*)d_in[8];
  const float* b_down = (const float*)d_in[9];
  const float* gpr    = (const float*)d_in[10];
  const float* w_ec   = (const float*)d_in[11];
  const float* b_ec   = (const float*)d_in[12];
  const float* bne_g  = (const float*)d_in[13];
  const float* bne_b  = (const float*)d_in[14];
  const float* bne_m  = (const float*)d_in[15];
  const float* bne_v  = (const float*)d_in[16];
  const float* w_fc2  = (const float*)d_in[17];
  const float* b_fc2  = (const float*)d_in[18];
  const float* bn2_g  = (const float*)d_in[19];
  const float* bn2_b  = (const float*)d_in[20];
  const float* bn2_m  = (const float*)d_in[21];
  const float* bn2_v  = (const float*)d_in[22];
  const float* w_up   = (const float*)d_in[23];
  const float* b_up   = (const float*)d_in[24];
  float* out = (float*)d_out;

  char* p = (char*)d_ws;
  auto alloc = [&](size_t bytes) -> char* {
    char* r = p; p += (bytes + 255) & ~(size_t)255; return r;
  };
  u16*   w1bf   = (u16*)  alloc((size_t)CC * CC * 2);
  float* bias1  = (float*)alloc(CC * 4);
  u16*   wdnbf  = (u16*)  alloc((size_t)RR * CC * 2);
  u16*   gptbf  = (u16*)  alloc((size_t)CC * RR * 2);
  u16*   wcomb  = (u16*)  alloc((size_t)2 * C2 * CC * 2);
  float* se     = (float*)alloc(C2 * 4);
  float* ce     = (float*)alloc(C2 * 4);
  u16*   w2bf   = (u16*)  alloc((size_t)CC * C2 * 2);
  float* bias2  = (float*)alloc(CC * 4);
  u16*   xT     = (u16*)  alloc((size_t)BB * NP * CC * 2);   // reused as xfT later? no: separate
  float* h      = (float*)alloc((size_t)BB * CC * NP * 4);
  u16*   hT     = (u16*)  alloc((size_t)BB * NP * CC * 2);
  float* lr     = (float*)alloc((size_t)BB * RR * NP * 4);
  u16*   lrT    = (u16*)  alloc((size_t)BB * NP * RR * 2);
  u16*   xfT    = (u16*)  alloc((size_t)BB * NP * CC * 2);
  float* nrm    = (float*)alloc((size_t)BB * NP * 4);
  float* G      = (float*)alloc((size_t)BB * NP * NP * 4);
  int*   nnidx  = (int*)  alloc((size_t)BB * NN * KNN * 4);
  float* Z      = (float*)alloc((size_t)BB * 2 * C2 * NP * 4);
  u16*   ymT    = (u16*)  alloc((size_t)BB * NP * C2 * 4 / 2);
  float* lrmT   = (float*)alloc((size_t)BB * NP * RR * 4);

  const int T = 256;
  k_prep_w1   <<<(CC*CC + T-1)/T, T, 0, stream>>>(w_fc1, b_fc1, bn1_g, bn1_b, bn1_m, bn1_v, w1bf, bias1);
  k_prep_small<<<(2*RR*CC + T-1)/T, T, 0, stream>>>(w_down, gpr, wdnbf, gptbf);
  k_prep_wcomb<<<(2*C2*CC + T-1)/T, T, 0, stream>>>(w_ec, b_ec, bne_g, bne_b, bne_m, bne_v, wcomb, se, ce);
  k_prep_w2   <<<(CC*C2 + T-1)/T, T, 0, stream>>>(w_fc2, b_fc2, bn2_g, bn2_b, bn2_m, bn2_v, w2bf, bias2);

  k_x_to_bf   <<<(BB*NP*CC)/T, T, 0, stream>>>(x, xT);
  k_gemm_fc1  <<<dim3(2, 7, BB), T, 0, stream>>>(w1bf, xT, bias1, h);
  k_prompt_pad<<<(BB*CC*28 + T-1)/T, T, 0, stream>>>(nprom, h);
  k_h_to_bf   <<<(BB*NP*CC)/T, T, 0, stream>>>(h, hT);

  k_gemm_down <<<dim3(1, 1, BB), T, 0, stream>>>(wdnbf, hT, b_down, lr, lrT);
  k_gemm_res  <<<dim3(2, 7, BB), T, 0, stream>>>(gptbf, lrT, h, xfT);

  k_norm      <<<(BB*NP)/8, T, 0, stream>>>(xfT, nrm);
  k_gemm_gram <<<dim3(1, 7, BB), T, 0, stream>>>(xfT, G);
  k_topk      <<<(BB*NN + T-1)/T, T, 0, stream>>>(G, nrm, nnidx);

  k_gemm_big  <<<dim3(6, 7, BB), T, 0, stream>>>(wcomb, xfT, Z);
  k_edge_max  <<<(BB*NP*C2)/T, T, 0, stream>>>(Z, nnidx, se, ce, ymT);
  k_lrmean    <<<(BB*NP*RR)/T, T, 0, stream>>>(lr, nnidx, lrmT);

  k_gemm_fc2  <<<dim3(2, 7, BB), T, 0, stream>>>(w2bf, ymT, bias2, w_up, b_up, lrmT, x, out);
}
